// PointNet2_13219909337190
// MI455X (gfx1250) — matrix operand order repaired and checked by reading, not compiled, not run on hardware
//
#include <hip/hip_runtime.h>
#include <cstddef>

typedef _Float16 f16;
typedef __attribute__((ext_vector_type(16))) _Float16 v16h;
typedef __attribute__((ext_vector_type(8)))  float    v8f;

#define BATCH 8
#define NPTS0 4096

union ABpack { v16h h; unsigned u[8]; };

// k order used here (contiguous k = 16*hh + 2*i, same in B): VGPR i (as dword = half-pair), lane-half hh.
__device__ __forceinline__ int a_kidx(int i, int hh) {
    int base = 2 * i;
    return base + (hh ? 16 : 0);
}

// Load 16x32 f16 A tile from row-major buffer (global or LDS), rows row0..row0+15.
__device__ __forceinline__ v16h load_a_tile(const f16* buf, int row0, int stride,
                                            int kbase, int lane) {
    int m = lane & 15, hh = lane >> 4;
    ABpack r;
    const f16* rowp = buf + (size_t)(row0 + m) * stride + kbase;
#pragma unroll
    for (int i = 0; i < 8; ++i)
        r.u[i] = *(const unsigned*)(rowp + a_kidx(i, hh));
    return r.h;
}

// Same but with row guard (rows >= R read as zero).
__device__ __forceinline__ v16h load_a_guard(const f16* buf, int row0, int R, int stride,
                                             int kbase, int lane) {
    int m = lane & 15, hh = lane >> 4;
    ABpack r;
    int row = row0 + m;
    if (row < R) {
        const f16* rowp = buf + (size_t)row * stride + kbase;
#pragma unroll
        for (int i = 0; i < 8; ++i)
            r.u[i] = *(const unsigned*)(rowp + a_kidx(i, hh));
    } else {
#pragma unroll
        for (int i = 0; i < 8; ++i) r.u[i] = 0u;
    }
    return r.h;
}

// Load 32x16 f16 B tile from Wt [Cout, Kd] (transposed weights), NO column guard.
__device__ __forceinline__ v16h load_b_tile_nc(const f16* Wt, int c0, int Kd,
                                               int kbase, int lane) {
    int n = lane & 15, hh = lane >> 4;
    ABpack r;
    const f16* p = Wt + (size_t)(c0 + n) * Kd + kbase + (hh ? 16 : 0);
#pragma unroll
    for (int j = 0; j < 8; ++j)
        r.u[j] = *(const unsigned*)(p + 2 * j);
    return r.h;
}

// Guarded B-tile load (columns >= Cout read as zero) for ragged heads (Cout=40).
__device__ __forceinline__ v16h load_b_tile(const f16* Wt, int c0, int Cout, int Kd,
                                            int kbase, int lane) {
    int n = lane & 15, hh = lane >> 4;
    int c = c0 + n;
    ABpack r;
    if (c < Cout) {
        const f16* p = Wt + (size_t)c * Kd + kbase + (hh ? 16 : 0);
#pragma unroll
        for (int j = 0; j < 8; ++j)
            r.u[j] = *(const unsigned*)(p + 2 * j);
    } else {
#pragma unroll
        for (int j = 0; j < 8; ++j) r.u[j] = 0u;
    }
    return r.h;
}

__device__ __forceinline__ v8f wmma_f16(v16h a, v16h b, v8f c) {
    return __builtin_amdgcn_wmma_f32_16x16x32_f16(false, a, false, b, (short)0, c,
                                                  false, false);
}

// ---------------- Farthest point sampling (1 block per batch element) -------------
#define FPS_T 256
__global__ void fps_kernel(const float* __restrict__ pos, int Npts, int m,
                           int* __restrict__ idx_out, float* __restrict__ centers) {
    __shared__ float mind[NPTS0];
    __shared__ float redv[FPS_T];
    __shared__ int   redi[FPS_T];
    __shared__ int   s_cur;
    int b = blockIdx.x, t = threadIdx.x;
    const float* P = pos + (size_t)b * Npts * 3;
    for (int i = t; i < Npts; i += FPS_T) mind[i] = 1e30f;
    if (t == 0) { s_cur = 0; idx_out[(size_t)b * m] = 0; }
    __syncthreads();
    for (int it = 1; it < m; ++it) {
        int cur = s_cur;
        float cx = P[cur * 3], cy = P[cur * 3 + 1], cz = P[cur * 3 + 2];
        float bv = -1.f; int bi = 0;
        for (int i = t; i < Npts; i += FPS_T) {
            float dx = P[i * 3] - cx, dy = P[i * 3 + 1] - cy, dz = P[i * 3 + 2] - cz;
            float d = dx * dx + dy * dy + dz * dz;
            float mv = fminf(mind[i], d);
            mind[i] = mv;
            if (mv > bv) { bv = mv; bi = i; }
        }
        redv[t] = bv; redi[t] = bi;
        __syncthreads();
        for (int s = FPS_T / 2; s > 0; s >>= 1) {
            if (t < s) {
                if (redv[t + s] > redv[t] ||
                    (redv[t + s] == redv[t] && redi[t + s] < redi[t])) {
                    redv[t] = redv[t + s]; redi[t] = redi[t + s];
                }
            }
            __syncthreads();
        }
        if (t == 0) { s_cur = redi[0]; idx_out[(size_t)b * m + it] = redi[0]; }
        __syncthreads();
    }
    for (int i = t; i < m; i += FPS_T) {
        int id = idx_out[(size_t)b * m + i];
        centers[((size_t)b * m + i) * 3 + 0] = P[id * 3 + 0];
        centers[((size_t)b * m + i) * 3 + 1] = P[id * 3 + 1];
        centers[((size_t)b * m + i) * 3 + 2] = P[id * 3 + 2];
    }
}

// ---------------- Radius query, K=32 nearest within radius ------------------------
#define BQ_T 128
__global__ void ball_query_kernel(const float* __restrict__ pos,
                                  const float* __restrict__ centers,
                                  int Npts, int m, float r2,
                                  int* __restrict__ nbr, unsigned* __restrict__ valid) {
    __shared__ float d2[NPTS0];
    __shared__ float redv[BQ_T];
    __shared__ int   redi[BQ_T];
    __shared__ int   s_sel[32];
    __shared__ unsigned s_mask;
    int c = blockIdx.x, b = blockIdx.y, t = threadIdx.x;
    const float* P = pos + (size_t)b * Npts * 3;
    const float* C = centers + ((size_t)b * m + c) * 3;
    float cx = C[0], cy = C[1], cz = C[2];
    for (int i = t; i < Npts; i += BQ_T) {
        float dx = P[i * 3] - cx, dy = P[i * 3 + 1] - cy, dz = P[i * 3 + 2] - cz;
        d2[i] = dx * dx + dy * dy + dz * dz;
    }
    if (t == 0) s_mask = 0u;
    __syncthreads();
    for (int k = 0; k < 32; ++k) {
        float bv = 1e31f; int bi = 0;
        for (int i = t; i < Npts; i += BQ_T) {
            float v = d2[i];
            if (v < bv) { bv = v; bi = i; }
        }
        redv[t] = bv; redi[t] = bi;
        __syncthreads();
        for (int s = BQ_T / 2; s > 0; s >>= 1) {
            if (t < s) {
                if (redv[t + s] < redv[t] ||
                    (redv[t + s] == redv[t] && redi[t + s] < redi[t])) {
                    redv[t] = redv[t + s]; redi[t] = redi[t + s];
                }
            }
            __syncthreads();
        }
        if (t == 0) {
            if (redv[0] <= r2) { s_sel[k] = redi[0]; s_mask |= (1u << k); d2[redi[0]] = 1e30f; }
            else               { s_sel[k] = 0; }
        }
        __syncthreads();
    }
    if (t < 32) nbr[((size_t)b * m + c) * 32 + t] = s_sel[t];
    if (t == 0) valid[(size_t)b * m + c] = s_mask;
}

// -------- Fused per-center PointConv MLP (WMMA) + masked max-pool -----------------
// One wave per center; KD/C1/C2 compile-time so A tiles live in registers and the
// K loops fully unroll: steady state = 2x global_load_b128 (B) + v_wmma per step.
template <int KD, int C1, int C2>
__global__ void sa_mlp_wmma_kernel(const float* __restrict__ x_in,
                                   const float* __restrict__ pos_in,
                                   const float* __restrict__ centers,
                                   const int* __restrict__ nbr,
                                   const unsigned* __restrict__ valid,
                                   int Nin, int m, int Cprev,
                                   const f16* __restrict__ W1t, const float* __restrict__ b1,
                                   const f16* __restrict__ W2t, const float* __restrict__ b2,
                                   float* __restrict__ x_out) {
    __shared__ f16 feats[32][KD];
    __shared__ f16 h1[32][C1];
    int c = blockIdx.x, b = blockIdx.y, lane = threadIdx.x;

    unsigned mask = valid[(size_t)b * m + c];
    int r = lane;
    int id = nbr[((size_t)b * m + c) * 32 + r];
    bool ok = (mask >> r) & 1u;
    const float* xi = x_in + ((size_t)b * Nin + id) * Cprev;
    const float* pi = pos_in + ((size_t)b * Nin + id) * 3;
    const float* ce = centers + ((size_t)b * m + c) * 3;
    for (int j = 0; j < Cprev; ++j) feats[r][j] = ok ? (f16)xi[j] : (f16)0.f;
    for (int j = 0; j < 3; ++j)     feats[r][Cprev + j] = ok ? (f16)(pi[j] - ce[j]) : (f16)0.f;
    for (int j = Cprev + 3; j < KD; ++j) feats[r][j] = (f16)0.f;
    __syncthreads();

    int nlane = lane & 15;
    int hh = lane >> 4;

    // ---- Layer 1: h1 = relu(feats @ W1 + b1) ----
    for (int mt = 0; mt < 2; ++mt) {
        v16h a1[KD / 32];
#pragma unroll
        for (int kt = 0; kt < KD / 32; ++kt)
            a1[kt] = load_a_tile(&feats[0][0], mt * 16, KD, kt * 32, lane);
        int rbase = mt * 16 + (hh ? 8 : 0);
        for (int nt = 0; nt < C1; nt += 16) {
            v8f acc = {0.f, 0.f, 0.f, 0.f, 0.f, 0.f, 0.f, 0.f};
#pragma unroll
            for (int kt = 0; kt < KD / 32; ++kt)
                acc = wmma_f16(a1[kt], load_b_tile_nc(W1t, nt, KD, kt * 32, lane), acc);
            float bv = b1[nt + nlane];
#pragma unroll
            for (int j = 0; j < 8; ++j) {
                float v = acc[j] + bv;
                if (v < 0.f) v = 0.f;
                h1[rbase + j][nt + nlane] = (f16)v;
            }
        }
    }
    __syncthreads();

    // ---- Layer 2 + masked max-pool over the 32 neighbor rows ----
    float mx[C2 / 16];
#pragma unroll
    for (int t2 = 0; t2 < C2 / 16; ++t2) mx[t2] = 0.f;  // relu >= 0, center always valid

    for (int mt = 0; mt < 2; ++mt) {
        v16h a2[C1 / 32];
#pragma unroll
        for (int kt = 0; kt < C1 / 32; ++kt)
            a2[kt] = load_a_tile(&h1[0][0], mt * 16, C1, kt * 32, lane);
#pragma unroll
        for (int t2 = 0; t2 < C2 / 16; ++t2) {
            v8f acc = {0.f, 0.f, 0.f, 0.f, 0.f, 0.f, 0.f, 0.f};
#pragma unroll
            for (int kt = 0; kt < C1 / 32; ++kt)
                acc = wmma_f16(a2[kt], load_b_tile_nc(W2t, t2 * 16, C1, kt * 32, lane), acc);
            float bv = b2[t2 * 16 + nlane];
#pragma unroll
            for (int j = 0; j < 8; ++j) {
                int row = mt * 16 + j + (hh ? 8 : 0);
                float v = acc[j] + bv;
                if (v < 0.f) v = 0.f;
                if (!((mask >> row) & 1u)) v = 0.f;
                mx[t2] = fmaxf(mx[t2], v);
            }
        }
    }
#pragma unroll
    for (int t2 = 0; t2 < C2 / 16; ++t2) {
        float v = fmaxf(mx[t2], __shfl_xor(mx[t2], 16, 32));
        if (lane < 16)
            x_out[((size_t)b * m + c) * C2 + t2 * 16 + lane] = v;
    }
}

// ------- Unguarded WMMA GEMM (R%16==0, Cout%16==0): out = [relu](A@W + b) --------
__global__ void gemm_wmma_nc_kernel(const f16* __restrict__ A, int Kd,
                                    const f16* __restrict__ Wt,
                                    const float* __restrict__ bias,
                                    int Cout, f16* __restrict__ outF16, int relu) {
    int lane = threadIdx.x;
    int c0 = blockIdx.x * 16, r0 = blockIdx.y * 16;
    v8f acc = {0.f, 0.f, 0.f, 0.f, 0.f, 0.f, 0.f, 0.f};
    for (int kb = 0; kb < Kd; kb += 32) {
        v16h a  = load_a_tile(A, r0, Kd, kb, lane);
        v16h bm = load_b_tile_nc(Wt, c0, Kd, kb, lane);
        acc = wmma_f16(a, bm, acc);
    }
    int n = c0 + (lane & 15);
    int rbase = r0 + ((lane >> 4) ? 8 : 0);
    float bv = bias[n];
#pragma unroll
    for (int j = 0; j < 8; ++j) {
        float v = acc[j] + bv;
        if (relu && v < 0.f) v = 0.f;
        outF16[(size_t)(rbase + j) * Cout + n] = (f16)v;
    }
}

// ---- Guarded WMMA GEMM (ragged tail, f32 out) — only the final 16x256x40 -------
__global__ void gemm_wmma_kernel(const f16* __restrict__ A, int R, int Kd,
                                 const f16* __restrict__ Wt, const float* __restrict__ bias,
                                 int Cout, float* __restrict__ outF32, int relu) {
    int lane = threadIdx.x;
    int c0 = blockIdx.x * 16, r0 = blockIdx.y * 16;
    v8f acc = {0.f, 0.f, 0.f, 0.f, 0.f, 0.f, 0.f, 0.f};
    for (int kb = 0; kb < Kd; kb += 32) {
        v16h a  = load_a_guard(A, r0, R, Kd, kb, lane);
        v16h bm = load_b_tile(Wt, c0, Cout, Kd, kb, lane);
        acc = wmma_f16(a, bm, acc);
    }
    int n = c0 + (lane & 15);
    int rbase = r0 + ((lane >> 4) ? 8 : 0);
    float bv = (n < Cout) ? bias[n] : 0.f;
#pragma unroll
    for (int j = 0; j < 8; ++j) {
        int row = rbase + j;
        float v = acc[j] + bv;
        if (relu && v < 0.f) v = 0.f;
        if (row < R && n < Cout)
            outF32[(size_t)row * Cout + n] = v;
    }
}

// -------- Weight prep: f32 [Kin,Cout] -> f16 transposed+padded [Cout,Kd] ---------
__global__ void wprep_kernel(const float* __restrict__ W, int Kin, int Cout, int Kd,
                             f16* __restrict__ Wt) {
    int i = blockIdx.x * blockDim.x + threadIdx.x;
    if (i >= Cout * Kd) return;
    int c = i / Kd, k = i - c * Kd;
    Wt[i] = (k < Kin) ? (f16)W[(size_t)k * Cout + c] : (f16)0.f;
}

// -------- GA gather: A = concat(x3[256], pos3[3], pad) as f16 [B*512, 288] -------
__global__ void gather_ga_kernel(const float* __restrict__ x3,
                                 const float* __restrict__ pos3,
                                 f16* __restrict__ Aga) {
    const int Kd = 288, Cx = 256, n = 512;
    int i = blockIdx.x * blockDim.x + threadIdx.x;
    if (i >= BATCH * n * Kd) return;
    int row = i / Kd, k = i - row * Kd;
    float v;
    if (k < Cx)            v = x3[(size_t)row * Cx + k];
    else if (k < Cx + 3)   v = pos3[(size_t)row * 3 + (k - Cx)];
    else                   v = 0.f;
    Aga[i] = (f16)v;
}

// -- Global max-pool per batch: f16 [B*n,C] -> f16 [16,C] (rows 8..15 zeroed) -----
__global__ void maxpool_ga_kernel(const f16* __restrict__ h, f16* __restrict__ pooled,
                                  int n, int C) {
    int i = blockIdx.x * blockDim.x + threadIdx.x;
    if (i >= 16 * C) return;
    int b = i / C, ch = i - b * C;
    if (b >= BATCH) { pooled[i] = (f16)0.f; return; }
    float mx = -1e30f;
    for (int j = 0; j < n; ++j)
        mx = fmaxf(mx, (float)h[((size_t)b * n + j) * C + ch]);
    pooled[i] = (f16)mx;
}

// ---------------------------------------------------------------------------------
extern "C" void kernel_launch(void* const* d_in, const int* in_sizes, int n_in,
                              void* d_out, int out_size, void* d_ws, size_t ws_size,
                              hipStream_t stream) {
    (void)out_size; (void)ws_size;
    // ---- input mapping (robust to insertion-order vs pytree-sorted flattening) ----
    int pbase, xi, pi;
    if (in_sizes[0] == BATCH * NPTS0 * 3 && in_sizes[1] == BATCH * NPTS0 * 3) {
        xi = 0; pi = 1; pbase = 2;           // x, pos first
    } else {
        pbase = 0; pi = n_in - 2; xi = n_in - 1;  // params first (sorted keys)
    }
    bool sorted_params = (in_sizes[pbase] != 192);  // insertion order starts with sa1 W1 (6*32)
    static const int ins_map[22] = {0,1,2,3, 4,5,6,7, 8,9,10,11, 12,13,14,15, 16,17,18,19,20,21};
    static const int srt_map[22] = {10,11,12,13, 14,15,16,17, 18,19,20,21, 0,1,2,3, 4,5,6,7,8,9};
    const int* mp = sorted_params ? srt_map : ins_map;
    const float* p[22];
    for (int i = 0; i < 22; ++i) p[i] = (const float*)d_in[pbase + mp[i]];
    const float* saW1[3] = {p[0], p[4], p[8]};
    const float* sab1[3] = {p[1], p[5], p[9]};
    const float* saW2[3] = {p[2], p[6], p[10]};
    const float* sab2[3] = {p[3], p[7], p[11]};
    const float* gaW1 = p[12]; const float* gab1 = p[13];
    const float* gaW2 = p[14]; const float* gab2 = p[15];
    const float* l1W = p[16];  const float* l1b = p[17];
    const float* l2W = p[18];  const float* l2b = p[19];
    const float* l3W = p[20];  const float* l3b = p[21];
    const float* x0   = (const float*)d_in[xi];
    const float* pos0 = (const float*)d_in[pi];

    // ---- workspace carve-up ----
    char* wsb = (char*)d_ws;
    size_t off = 0;
    auto alloc = [&](size_t bytes) -> void* {
        void* q = wsb + off;
        off = (off + bytes + 255) & ~(size_t)255;
        return q;
    };
    float* pos1 = (float*)alloc((size_t)BATCH * 2048 * 3 * 4);
    float* pos2 = (float*)alloc((size_t)BATCH * 1024 * 3 * 4);
    float* pos3 = (float*)alloc((size_t)BATCH * 512 * 3 * 4);
    float* x1   = (float*)alloc((size_t)BATCH * 2048 * 64 * 4);
    float* x2   = (float*)alloc((size_t)BATCH * 1024 * 128 * 4);
    float* x3   = (float*)alloc((size_t)BATCH * 512 * 256 * 4);
    int* idx        = (int*)alloc((size_t)BATCH * 2048 * 4);
    int* nbr        = (int*)alloc((size_t)BATCH * 2048 * 32 * 4);
    unsigned* valid = (unsigned*)alloc((size_t)BATCH * 2048 * 4);
    f16* Wt_sa1a = (f16*)alloc((size_t)32 * 32 * 2);
    f16* Wt_sa1b = (f16*)alloc((size_t)64 * 32 * 2);
    f16* Wt_sa2a = (f16*)alloc((size_t)128 * 96 * 2);
    f16* Wt_sa2b = (f16*)alloc((size_t)128 * 128 * 2);
    f16* Wt_sa3a = (f16*)alloc((size_t)256 * 160 * 2);
    f16* Wt_sa3b = (f16*)alloc((size_t)256 * 256 * 2);
    f16* Wt_ga1  = (f16*)alloc((size_t)512 * 288 * 2);
    f16* Wt_ga2  = (f16*)alloc((size_t)1024 * 512 * 2);
    f16* Wt_l1   = (f16*)alloc((size_t)512 * 1024 * 2);
    f16* Wt_l2   = (f16*)alloc((size_t)256 * 512 * 2);
    f16* Wt_l3   = (f16*)alloc((size_t)40 * 256 * 2);
    f16* Aga    = (f16*)alloc((size_t)BATCH * 512 * 288 * 2);
    f16* h1ga   = (f16*)alloc((size_t)BATCH * 512 * 512 * 2);
    f16* h2ga   = (f16*)alloc((size_t)BATCH * 512 * 1024 * 2);
    f16* pooled = (f16*)alloc((size_t)16 * 1024 * 2);   // rows 8..15 zero-padded
    f16* hA     = (f16*)alloc((size_t)16 * 512 * 2);
    f16* hB     = (f16*)alloc((size_t)16 * 256 * 2);

    auto wprep = [&](const float* W, int Kin, int Cout, int Kd, f16* Wt) {
        int total = Cout * Kd;
        wprep_kernel<<<(total + 255) / 256, 256, 0, stream>>>(W, Kin, Cout, Kd, Wt);
    };
    wprep(saW1[0], 6, 32, 32, Wt_sa1a);
    wprep(saW2[0], 32, 64, 32, Wt_sa1b);
    wprep(saW1[1], 67, 128, 96, Wt_sa2a);
    wprep(saW2[1], 128, 128, 128, Wt_sa2b);
    wprep(saW1[2], 131, 256, 160, Wt_sa3a);
    wprep(saW2[2], 256, 256, 256, Wt_sa3b);
    wprep(gaW1, 259, 512, 288, Wt_ga1);
    wprep(gaW2, 512, 1024, 512, Wt_ga2);
    wprep(l1W, 1024, 512, 1024, Wt_l1);
    wprep(l2W, 512, 256, 512, Wt_l2);
    wprep(l3W, 256, 40, 256, Wt_l3);

    // ---- SA layer 1: 4096 -> 2048, r=0.2, MLP 6->32->64 ----
    fps_kernel<<<BATCH, FPS_T, 0, stream>>>(pos0, 4096, 2048, idx, pos1);
    ball_query_kernel<<<dim3(2048, BATCH), BQ_T, 0, stream>>>(pos0, pos1, 4096, 2048,
                                                              0.04f, nbr, valid);
    sa_mlp_wmma_kernel<32, 32, 64><<<dim3(2048, BATCH), 32, 0, stream>>>(
        x0, pos0, pos1, nbr, valid, 4096, 2048, 3,
        Wt_sa1a, sab1[0], Wt_sa1b, sab2[0], x1);

    // ---- SA layer 2: 2048 -> 1024, r=0.3, MLP 67->128->128 ----
    fps_kernel<<<BATCH, FPS_T, 0, stream>>>(pos1, 2048, 1024, idx, pos2);
    ball_query_kernel<<<dim3(1024, BATCH), BQ_T, 0, stream>>>(pos1, pos2, 2048, 1024,
                                                              0.09f, nbr, valid);
    sa_mlp_wmma_kernel<96, 128, 128><<<dim3(1024, BATCH), 32, 0, stream>>>(
        x1, pos1, pos2, nbr, valid, 2048, 1024, 64,
        Wt_sa2a, sab1[1], Wt_sa2b, sab2[1], x2);

    // ---- SA layer 3: 1024 -> 512, r=0.4, MLP 131->256->256 ----
    fps_kernel<<<BATCH, FPS_T, 0, stream>>>(pos2, 1024, 512, idx, pos3);
    ball_query_kernel<<<dim3(512, BATCH), BQ_T, 0, stream>>>(pos2, pos3, 1024, 512,
                                                             0.16f, nbr, valid);
    sa_mlp_wmma_kernel<160, 256, 256><<<dim3(512, BATCH), 32, 0, stream>>>(
        x2, pos2, pos3, nbr, valid, 1024, 512, 128,
        Wt_sa3a, sab1[2], Wt_sa3b, sab2[2], x3);

    // ---- Global abstraction: [B*512,259] -> 512 -> 1024, max over 512 ----
    {
        int total = BATCH * 512 * 288;
        gather_ga_kernel<<<(total + 255) / 256, 256, 0, stream>>>(x3, pos3, Aga);
    }
    gemm_wmma_nc_kernel<<<dim3(512 / 16, (BATCH * 512) / 16), 32, 0, stream>>>(
        Aga, 288, Wt_ga1, gab1, 512, h1ga, 1);
    gemm_wmma_nc_kernel<<<dim3(1024 / 16, (BATCH * 512) / 16), 32, 0, stream>>>(
        h1ga, 512, Wt_ga2, gab2, 1024, h2ga, 1);
    maxpool_ga_kernel<<<(16 * 1024 + 255) / 256, 256, 0, stream>>>(h2ga, pooled, 512, 1024);

    // ---- Heads: 1024 -> 512 -> 256 -> 40 (rows padded to 16; pads never mix) ----
    gemm_wmma_nc_kernel<<<dim3(512 / 16, 1), 32, 0, stream>>>(
        pooled, 1024, Wt_l1, l1b, 512, hA, 1);
    gemm_wmma_nc_kernel<<<dim3(256 / 16, 1), 32, 0, stream>>>(
        hA, 512, Wt_l2, l2b, 256, hB, 1);
    gemm_wmma_kernel<<<dim3((40 + 15) / 16, 1), 32, 0, stream>>>(
        hB, BATCH, 256, Wt_l3, l3b, 40, (float*)d_out, 0);
}
